// TransformerBlock_74328704024931
// MI455X (gfx1250) — matrix operand order repaired and checked by reading, not compiled, not run on hardware
//
#include <hip/hip_runtime.h>
#include <hip/hip_fp16.h>

// ---------------------------------------------------------------------------
// GPT-2 transformer block forward for MI455X (gfx1250), wave32 + WMMA.
//   D_MODEL=768, N_HEADS=12, D_HEAD=64, D_MLP=3072, B=2, S=2048
// All GEMMs run on v_wmma_f32_16x16x32_f16 (f16 in, f32 accumulate).
// GEMM: 128x64 block tile, 8 waves, 32x32 wave tile (4 WMMA/K-step with
// fragment reuse), LDS double-buffering via GLOBAL_LOAD_ASYNC_TO_LDS_B128
// (ASYNCcnt) + global_prefetch_b8 for the K+64 tile.
// ---------------------------------------------------------------------------

#define D_MODEL 768
#define N_HEADS 12
#define D_HEAD  64
#define D_MLP   3072
#define BATCH   2
#define SEQ     2048
#define NTOK    (BATCH * SEQ)      // 4096 rows
#define QKV_N   (3 * D_MODEL)      // 2304 fused QKV columns
#define LN_EPS  1e-5f

typedef __attribute__((ext_vector_type(16))) _Float16 v16h;
typedef __attribute__((ext_vector_type(8)))  float    v8f;

// ---------------------------------------------------------------------------
// Async global -> LDS copy of 16 bytes per lane (CDNA5 async-DMA path,
// CDNA5 ISA). VDST = 32-bit LDS byte address (LDS
// address space is 32-bit LDS-relative on AMDGPU), VADDR = 64-bit global
// address, GV mode (saddr = off). Tracked by ASYNCcnt.
// ---------------------------------------------------------------------------
__device__ __forceinline__ void async_copy16(void* lds_ptr, const void* gptr) {
  unsigned lds_off = (unsigned)(unsigned long long)
      (__attribute__((address_space(3))) void*)lds_ptr;
  asm volatile("global_load_async_to_lds_b128 %0, %1, off"
               :: "v"(lds_off), "v"((unsigned long long)gptr)
               : "memory");
}
__device__ __forceinline__ void wait_async0() {
  asm volatile("s_wait_asynccnt 0" ::: "memory");
}

// ---------------------------------------------------------------------------
// Elementwise f32 -> f16 cast (weights already in [K,N] row-major order).
// ---------------------------------------------------------------------------
__global__ __launch_bounds__(256) void cvt_f16_kernel(const float* __restrict__ x,
                                                      _Float16* __restrict__ y, int n) {
  int i = blockIdx.x * 256 + threadIdx.x;
  if (i < n) y[i] = (_Float16)x[i];
}

// ---------------------------------------------------------------------------
// Pack W_Q/W_K/W_V (n_heads, d_model, d_head) into fused f16 Wqkv[768][2304]
// with column layout  [ Q(n*64+h) | K(n*64+h) | V(n*64+h) ], plus fused bias.
// ---------------------------------------------------------------------------
__global__ __launch_bounds__(256) void pack_qkv_kernel(
    const float* __restrict__ WQ, const float* __restrict__ WK, const float* __restrict__ WV,
    const float* __restrict__ bQ, const float* __restrict__ bK, const float* __restrict__ bV,
    _Float16* __restrict__ Wqkv, float* __restrict__ bqkv) {
  int idx = blockIdx.x * 256 + threadIdx.x;
  if (idx < D_MODEL * QKV_N) {
    int e = idx / QKV_N;
    int col = idx % QKV_N;
    int which = col / D_MODEL;
    int nh = col % D_MODEL;            // n*64 + h
    int n = nh >> 6, h = nh & 63;
    const float* W = (which == 0) ? WQ : (which == 1) ? WK : WV;
    Wqkv[idx] = (_Float16)W[((size_t)n * D_MODEL + e) * D_HEAD + h];
  }
  if (idx < QKV_N) {
    int which = idx / D_MODEL, nh = idx % D_MODEL;
    const float* bb = (which == 0) ? bQ : (which == 1) ? bK : bV;
    bqkv[idx] = bb[nh];
  }
}

// ---------------------------------------------------------------------------
// LayerNorm: one 256-thread block per token row, f16 output for WMMA consumers.
// ---------------------------------------------------------------------------
__global__ __launch_bounds__(256) void ln_kernel(const float* __restrict__ x,
                                                 const float* __restrict__ w,
                                                 const float* __restrict__ b,
                                                 _Float16* __restrict__ y, int n) {
  int row = blockIdx.x, tid = threadIdx.x;
  const float* xr = x + (size_t)row * n;
  float s = 0.f, s2 = 0.f;
  for (int c = tid; c < n; c += 256) { float v = xr[c]; s += v; s2 += v * v; }
  __shared__ float r1[256], r2[256];
  r1[tid] = s; r2[tid] = s2;
  __syncthreads();
  for (int off = 128; off > 0; off >>= 1) {
    if (tid < off) { r1[tid] += r1[tid + off]; r2[tid] += r2[tid + off]; }
    __syncthreads();
  }
  float mean = r1[0] / (float)n;
  float var  = r2[0] / (float)n - mean * mean;
  float inv  = rsqrtf(var + LN_EPS);
  for (int c = tid; c < n; c += 256)
    y[(size_t)row * n + c] = (_Float16)((xr[c] - mean) * inv * w[c] + b[c]);
}

// ---------------------------------------------------------------------------
// GPT-2 "new" gelu
// ---------------------------------------------------------------------------
__device__ __forceinline__ float gelu_new(float x) {
  const float c = 0.7978845608028654f;  // sqrt(2/pi)
  return 0.5f * x * (1.0f + tanhf(c * (x + 0.044715f * x * x * x)));
}

// ---------------------------------------------------------------------------
// Tiled WMMA GEMM:  C[M,N] = A[M,K] (f16, row-major) * B[K,N] (f16, row-major)
// Block: 256 threads = 8 waves in a 4(M) x 2(N) grid -> 128x64 block tile.
// Each wave owns a 32x32 wave tile: 2 A-frags x 2 B-frags -> 4 WMMA per
// 32-wide K-step with operand reuse. LDS tiles are double-buffered with
// async global->LDS DMA: the next stage is issued right after the barrier
// and overlaps the current stage's WMMAs; each wave drains its own ASYNCcnt
// before the next barrier. K+64 tile warmed into L2 via global_prefetch_b8.
// Fragment layouts per CDNA5 ISA §7.12.2 (wave32, 16-bit A/B;
// 32-bit C/D: element r of lane l -> M = r + 8*(l>=16), N = l&15).
// ---------------------------------------------------------------------------
template <bool GELU, bool RESID, bool OUTH>
__global__ __launch_bounds__(256) void gemm_wmma_kernel(
    const _Float16* __restrict__ A, const _Float16* __restrict__ Bm,
    const float* __restrict__ bias, const float* __restrict__ resid,
    _Float16* __restrict__ out_h, float* __restrict__ out_f,
    int M, int N, int K) {
  __shared__ _Float16 As[2][128][40];  // 128 rows x 32 K (+8 pad, 16B-aligned rows)
  __shared__ _Float16 Bs[2][32][72];   // 32 K x 64 N (+8 pad)

  const int tid  = threadIdx.x;
  const int wave = tid >> 5;
  const int lane = tid & 31;
  const int l    = lane & 15;
  const int hi   = lane >> 4;
  const int wm   = wave >> 1;      // 0..3  (M direction)
  const int wn   = wave & 1;       // 0..1  (N direction)
  const int m0   = blockIdx.y * 128;
  const int n0   = blockIdx.x * 64;

  v8f acc[2][2];
#pragma unroll
  for (int mi = 0; mi < 2; ++mi)
#pragma unroll
    for (int ni = 0; ni < 2; ++ni)
#pragma unroll
      for (int r = 0; r < 8; ++r) acc[mi][ni][r] = 0.0f;

  // Staging assignments: A = 128x32 halves (16/thread), B = 32x64 halves (8/thread).
  const int ar  = tid >> 1;            // 0..127
  const int akc = (tid & 1) * 16;      // 0 or 16
  const int br  = tid >> 3;            // 0..31
  const int bc  = (tid & 7) * 8;       // 0..56

  auto stage = [&](int buf, int k0) {
    const _Float16* ap = A + (size_t)(m0 + ar) * K + (k0 + akc);
    async_copy16(&As[buf][ar][akc], ap);
    async_copy16(&As[buf][ar][akc + 8], ap + 8);
    const _Float16* bp = Bm + (size_t)(k0 + br) * N + (n0 + bc);
    async_copy16(&Bs[buf][br][bc], bp);
  };

  stage(0, 0);
  int cur = 0;
  for (int k0 = 0; k0 < K; k0 += 32) {
    wait_async0();                         // this wave's async stage complete
    __syncthreads();                       // all waves' stages visible
    if (k0 + 32 < K) {
      stage(cur ^ 1, k0 + 32);             // async DMA overlaps WMMAs below
      if (k0 + 64 < K) {                   // pull K+64 tile toward L2
        __builtin_prefetch(A + (size_t)(m0 + ar) * K + (k0 + 64 + akc), 0, 1);
        __builtin_prefetch(Bm + (size_t)(k0 + 64 + br) * N + (n0 + bc), 0, 1);
      }
    }

    // A fragments (16x32 f16): lane l<16 row=l holds K {0..7,16..23},
    // lane l+16 same row holds K {8..15,24..31}.
    v16h af[2];
    const int kb   = hi * 8;
    const int kb16 = hi * 16;
#pragma unroll
    for (int mi = 0; mi < 2; ++mi) {
      const int arow = wm * 32 + mi * 16 + l;
#pragma unroll
      for (int i = 0; i < 8; ++i) {
        int kk = ((i & 4) ? 16 : 0) + kb + 2 * (i & 3);
        af[mi][2 * i]     = As[cur][arow][kk];
        af[mi][2 * i + 1] = As[cur][arow][kk + 1];
      }
    }
    // B fragments (32x16 f16): lanes 0-15 col=l K {0..7,16..23}, lanes 16-31 K {8..15,24..31}.
    v16h bf[2];
#pragma unroll
    for (int ni = 0; ni < 2; ++ni) {
      const int bcol = wn * 32 + ni * 16 + l;
#pragma unroll
      for (int j = 0; j < 16; ++j) bf[ni][j] = Bs[cur][kb + j + (j & 8)][bcol];
    }
#pragma unroll
    for (int mi = 0; mi < 2; ++mi)
#pragma unroll
      for (int ni = 0; ni < 2; ++ni)
        acc[mi][ni] = __builtin_amdgcn_wmma_f32_16x16x32_f16(
            false, af[mi], false, bf[ni], (short)0, acc[mi][ni], false, false);
    cur ^= 1;
  }

  // Epilogue: bias / gelu / residual / store (f16 or f32).
#pragma unroll
  for (int ni = 0; ni < 2; ++ni) {
    const int col = n0 + wn * 32 + ni * 16 + l;
    const float bv = bias[col];
#pragma unroll
    for (int mi = 0; mi < 2; ++mi) {
#pragma unroll
      for (int r = 0; r < 8; ++r) {
        const int row = m0 + wm * 32 + mi * 16 + 8 * hi + r;
        float v = acc[mi][ni][r] + bv;
        if (GELU)  v = gelu_new(v);
        if (RESID) v += resid[(size_t)row * N + col];
        if (OUTH)  out_h[(size_t)row * N + col] = (_Float16)v;
        else       out_f[(size_t)row * N + col] = v;
      }
    }
  }
}

// ---------------------------------------------------------------------------
// Causal flash attention over fused qkv[NTOK][2304] (f16):
//   cols [0,768)=Q, [768,1536)=K, [1536,2304)=V, per-head 64-wide slices.
// Grid: (SEQ/64, N_HEADS, BATCH), 128 threads (4 waves); wave w owns queries
// [qblk*64 + w*16, +16). Keys/values stream through LDS (async DMA) in
// blocks of 32. Output z[NTOK][768] f16 with column n*64+h (W_O row order).
// ---------------------------------------------------------------------------
__global__ __launch_bounds__(128) void attn_kernel(const _Float16* __restrict__ qkv,
                                                   _Float16* __restrict__ z) {
  const int qblk  = blockIdx.x;
  const int head  = blockIdx.y;
  const int batch = blockIdx.z;
  const int tid   = threadIdx.x;
  const int wave  = tid >> 5;
  const int lane  = tid & 31;
  const int l     = lane & 15;
  const int hi    = lane >> 4;

  __shared__ _Float16 Ks[32][72];      // 32 keys x 64 feat (+8 pad)
  __shared__ _Float16 Vs[32][72];
  __shared__ _Float16 Ps[4][16][32];   // per-wave probability tile

  const int rowbase = batch * SEQ + qblk * 64;   // global token row of this block

  // Load the wave's Q tile (16x64) straight into two A fragments (K 0..31, 32..63).
  v16h qa0, qa1;
  {
    const _Float16* qp = qkv + (size_t)(rowbase + wave * 16 + l) * QKV_N + head * D_HEAD;
    const int kb = hi * 8;
#pragma unroll
    for (int i = 0; i < 8; ++i) {
      int kk = ((i & 4) ? 16 : 0) + kb + 2 * (i & 3);
      qa0[2 * i]     = qp[kk];
      qa0[2 * i + 1] = qp[kk + 1];
      qa1[2 * i]     = qp[32 + kk];
      qa1[2 * i + 1] = qp[32 + kk + 1];
    }
  }

  v8f o[4];
  float mrow[8], lrow[8];
#pragma unroll
  for (int t = 0; t < 4; ++t)
#pragma unroll
    for (int r = 0; r < 8; ++r) o[t][r] = 0.0f;
#pragma unroll
  for (int r = 0; r < 8; ++r) { mrow[r] = -1.0e30f; lrow[r] = 0.0f; }

  const int stg_r  = tid >> 2;         // 0..31
  const int stg_c0 = (tid & 3) * 16;   // 0,16,32,48
  const int kend   = qblk * 64 + 64;   // causal: keys in [0, kend)
  const int qmax   = qblk * 64 + wave * 16 + 15;

  for (int j0 = 0; j0 < kend; j0 += 32) {
    // Stage 32 K rows + 32 V rows via async global->LDS DMA (16B x4 per thread).
    {
      const size_t trow = (size_t)(batch * SEQ + j0 + stg_r) * QKV_N;
      const _Float16* kp = qkv + trow + D_MODEL + head * D_HEAD + stg_c0;
      const _Float16* vp = qkv + trow + 2 * D_MODEL + head * D_HEAD + stg_c0;
      async_copy16(&Ks[stg_r][stg_c0],     kp);
      async_copy16(&Ks[stg_r][stg_c0 + 8], kp + 8);
      async_copy16(&Vs[stg_r][stg_c0],     vp);
      async_copy16(&Vs[stg_r][stg_c0 + 8], vp + 8);
      if (j0 + 32 < kend) {              // warm L2 for the next key block
        __builtin_prefetch(kp + 32 * QKV_N, 0, 1);
        __builtin_prefetch(vp + 32 * QKV_N, 0, 1);
      }
    }
    wait_async0();
    __syncthreads();

    if (j0 <= qmax) {   // uniform per wave: this wave has unmasked keys here
      // ---- S = Q * K^T : two 16-key subtiles, K-dim 64 = 2 WMMA steps each.
      v8f s0, s1;
#pragma unroll
      for (int r = 0; r < 8; ++r) { s0[r] = 0.0f; s1[r] = 0.0f; }
      const int kb16 = hi * 16;
#pragma unroll
      for (int s = 0; s < 2; ++s) {
        v16h bf0, bf1;
#pragma unroll
        for (int j = 0; j < 16; ++j) {
          bf0[j] = Ks[l][s * 32 + hi * 8 + j + (j & 8)];        // key = l      (subtile 0)
          bf1[j] = Ks[16 + l][s * 32 + hi * 8 + j + (j & 8)];   // key = 16 + l (subtile 1)
        }
        const v16h aq = (s == 0) ? qa0 : qa1;
        s0 = __builtin_amdgcn_wmma_f32_16x16x32_f16(false, aq, false, bf0,
                                                    (short)0, s0, false, false);
        s1 = __builtin_amdgcn_wmma_f32_16x16x32_f16(false, aq, false, bf1,
                                                    (short)0, s1, false, false);
      }

      // ---- scale, causal mask, online softmax (row groups = 16-lane halves)
      float corr[8];
      const int qbase = qblk * 64 + wave * 16 + 8 * hi;
#pragma unroll
      for (int r = 0; r < 8; ++r) {
        const int qp  = qbase + r;
        const int kp0 = j0 + l;
        const int kp1 = j0 + 16 + l;
        float v0 = (kp0 <= qp) ? s0[r] * 0.125f : -1.0e30f;
        float v1 = (kp1 <= qp) ? s1[r] * 0.125f : -1.0e30f;
        float mx = fmaxf(v0, v1);
#pragma unroll
        for (int off = 8; off > 0; off >>= 1) mx = fmaxf(mx, __shfl_xor(mx, off, 16));
        const float mnew = fmaxf(mrow[r], mx);
        const float p0 = __expf(v0 - mnew);
        const float p1 = __expf(v1 - mnew);
        float rs = p0 + p1;
#pragma unroll
        for (int off = 8; off > 0; off >>= 1) rs += __shfl_xor(rs, off, 16);
        corr[r] = __expf(mrow[r] - mnew);
        lrow[r] = lrow[r] * corr[r] + rs;
        mrow[r] = mnew;
        Ps[wave][8 * hi + r][l]      = (_Float16)p0;
        Ps[wave][8 * hi + r][16 + l] = (_Float16)p1;
      }
#pragma unroll
      for (int t = 0; t < 4; ++t)
#pragma unroll
        for (int r = 0; r < 8; ++r) o[t][r] *= corr[r];

      // Per-wave LDS round-trip: wait only on the DS counter (wave-local).
      asm volatile("s_wait_dscnt 0" ::: "memory");

      // ---- O += P * V : P (16x32) as A fragment, 4 output subtiles of 16.
      v16h pa;
      {
        const int kb = hi * 8;
#pragma unroll
        for (int i = 0; i < 8; ++i) {
          int kk = ((i & 4) ? 16 : 0) + kb + 2 * (i & 3);
          pa[2 * i]     = Ps[wave][l][kk];
          pa[2 * i + 1] = Ps[wave][l][kk + 1];
        }
      }
#pragma unroll
      for (int t = 0; t < 4; ++t) {
        v16h bv;
#pragma unroll
        for (int j = 0; j < 16; ++j) bv[j] = Vs[hi * 8 + j + (j & 8)][t * 16 + l];
        o[t] = __builtin_amdgcn_wmma_f32_16x16x32_f16(false, pa, false, bv,
                                                      (short)0, o[t], false, false);
      }
    }
    __syncthreads();   // Ks/Vs reused next iteration
  }

  // ---- finalize: divide by row sums, store z[row][head*64 + col]
#pragma unroll
  for (int t = 0; t < 4; ++t) {
#pragma unroll
    for (int r = 0; r < 8; ++r) {
      const int row = rowbase + wave * 16 + 8 * hi + r;
      const float val = o[t][r] / lrow[r];
      z[(size_t)row * D_MODEL + head * D_HEAD + t * 16 + l] = (_Float16)val;
    }
  }
}

// ---------------------------------------------------------------------------
// Host-side orchestration (graph-capture safe: only kernel launches on stream).
// ---------------------------------------------------------------------------
extern "C" void kernel_launch(void* const* d_in, const int* in_sizes, int n_in,
                              void* d_out, int out_size, void* d_ws, size_t ws_size,
                              hipStream_t stream) {
  (void)in_sizes; (void)n_in; (void)out_size; (void)ws_size;
  const float* resid_pre = (const float*)d_in[0];
  const float* W_Q   = (const float*)d_in[1];
  const float* b_Q   = (const float*)d_in[2];
  const float* W_K   = (const float*)d_in[3];
  const float* b_K   = (const float*)d_in[4];
  const float* W_V   = (const float*)d_in[5];
  const float* b_V   = (const float*)d_in[6];
  const float* W_O   = (const float*)d_in[7];
  const float* b_O   = (const float*)d_in[8];
  const float* ln1_w = (const float*)d_in[9];
  const float* ln1_b = (const float*)d_in[10];
  const float* ln2_w = (const float*)d_in[11];
  const float* ln2_b = (const float*)d_in[12];
  const float* W_in  = (const float*)d_in[13];
  const float* b_in  = (const float*)d_in[14];
  const float* W_out = (const float*)d_in[15];
  const float* b_out = (const float*)d_in[16];
  float* out = (float*)d_out;

  // Workspace carve-out (bump allocator, 256B aligned). Total ~90 MB.
  char* p = (char*)d_ws;
  auto carve = [&](size_t bytes) -> void* {
    void* r = (void*)p;
    p += (bytes + 255) & ~(size_t)255;
    return r;
  };
  _Float16* xln1   = (_Float16*)carve((size_t)NTOK * D_MODEL * 2);
  _Float16* Wqkv   = (_Float16*)carve((size_t)D_MODEL * QKV_N * 2);
  float*    bqkv   = (float*)   carve((size_t)QKV_N * 4);
  _Float16* qkv    = (_Float16*)carve((size_t)NTOK * QKV_N * 2);
  _Float16* zbuf   = (_Float16*)carve((size_t)NTOK * D_MODEL * 2);
  _Float16* WOh    = (_Float16*)carve((size_t)D_MODEL * D_MODEL * 2);
  float*    rmid   = (float*)   carve((size_t)NTOK * D_MODEL * 4);
  _Float16* xln2   = (_Float16*)carve((size_t)NTOK * D_MODEL * 2);
  _Float16* Winh   = (_Float16*)carve((size_t)D_MODEL * D_MLP * 2);
  _Float16* hbuf   = (_Float16*)carve((size_t)NTOK * D_MLP * 2);
  _Float16* Wouth  = (_Float16*)carve((size_t)D_MLP * D_MODEL * 2);

  // ---- weight prep (f16)
  {
    int n = D_MODEL * D_MODEL;  // W_O flat (n*64+h, e) is already [K=768][N=768]
    cvt_f16_kernel<<<(n + 255) / 256, 256, 0, stream>>>(W_O, WOh, n);
    n = D_MODEL * D_MLP;
    cvt_f16_kernel<<<(n + 255) / 256, 256, 0, stream>>>(W_in, Winh, n);
    n = D_MLP * D_MODEL;
    cvt_f16_kernel<<<(n + 255) / 256, 256, 0, stream>>>(W_out, Wouth, n);
    n = D_MODEL * QKV_N;
    pack_qkv_kernel<<<(n + 255) / 256, 256, 0, stream>>>(W_Q, W_K, W_V, b_Q, b_K, b_V,
                                                         Wqkv, bqkv);
  }

  // ---- ln1
  ln_kernel<<<NTOK, 256, 0, stream>>>(resid_pre, ln1_w, ln1_b, xln1, D_MODEL);

  // ---- fused QKV projection: [4096,768] x [768,2304] -> qkv (f16)
  gemm_wmma_kernel<false, false, true>
      <<<dim3(QKV_N / 64, NTOK / 128), 256, 0, stream>>>(
          xln1, Wqkv, bqkv, nullptr, qkv, nullptr, NTOK, QKV_N, D_MODEL);

  // ---- causal flash attention -> z (f16, col = n*64+h)
  attn_kernel<<<dim3(SEQ / 64, N_HEADS, BATCH), 128, 0, stream>>>(qkv, zbuf);

  // ---- output projection + residual: rmid = z @ W_O + b_O + resid_pre (f32)
  gemm_wmma_kernel<false, true, false>
      <<<dim3(D_MODEL / 64, NTOK / 128), 256, 0, stream>>>(
          zbuf, WOh, b_O, resid_pre, nullptr, rmid, NTOK, D_MODEL, D_MODEL);

  // ---- ln2
  ln_kernel<<<NTOK, 256, 0, stream>>>(rmid, ln2_w, ln2_b, xln2, D_MODEL);

  // ---- MLP in + gelu: h = gelu(xln2 @ W_in + b_in) (f16)
  gemm_wmma_kernel<true, false, true>
      <<<dim3(D_MLP / 64, NTOK / 128), 256, 0, stream>>>(
          xln2, Winh, b_in, nullptr, hbuf, nullptr, NTOK, D_MLP, D_MODEL);

  // ---- MLP out + residual: out = h @ W_out + b_out + rmid (f32)
  gemm_wmma_kernel<false, true, false>
      <<<dim3(D_MODEL / 64, NTOK / 128), 256, 0, stream>>>(
          hbuf, Wouth, b_out, rmid, nullptr, out, NTOK, D_MLP, D_MODEL);
}
